// PointNet2Encoder_66520453480973
// MI455X (gfx1250) — compile-verified, matrix operand order checked by reading, not run on hardware
//
#include <hip/hip_runtime.h>
#include <stdint.h>

typedef __attribute__((ext_vector_type(16))) __bf16 v16bf;
typedef __attribute__((ext_vector_type(8)))  float  v8f;

__device__ __forceinline__ unsigned short f2bf(float f){
  unsigned u = __float_as_uint(f);
  u += 0x7FFFu + ((u >> 16) & 1u);            // round-to-nearest-even
  return (unsigned short)(u >> 16);
}

// ------------------------------------------------------------------
// convert f32 [rows,K] -> bf16 [rows,Kpad] zero-padded
// ------------------------------------------------------------------
__global__ void convert_pad_bf16_kernel(const float* __restrict__ src,
                                        unsigned short* __restrict__ dst,
                                        int rows, int K, int Kpad){
  long long t = (long long)blockIdx.x * blockDim.x + threadIdx.x;
  long long total = (long long)rows * Kpad;
  if (t >= total) return;
  int c = (int)(t % Kpad);
  long long r = t / Kpad;
  dst[t] = (c < K) ? f2bf(src[r * (long long)K + c]) : (unsigned short)0;
}

__global__ void zero_f32_kernel(float* __restrict__ p, int n){
  int t = blockIdx.x * blockDim.x + threadIdx.x;
  if (t < n) p[t] = 0.0f;
}

// ------------------------------------------------------------------
// Farthest point sampling: one block per batch, iterative argmax.
// ------------------------------------------------------------------
template<int PPT>
__global__ void fps_kernel(const float* __restrict__ xyz,
                           float* __restrict__ new_xyz,
                           int N, int npoint){
  const int b   = blockIdx.x;
  const int tid = threadIdx.x;
  const int T   = blockDim.x;                 // 256
  const float* X = xyz + (long long)b * N * 3;

  float dist[PPT];
#pragma unroll
  for (int i = 0; i < PPT; ++i) dist[i] = 1e10f;

  __shared__ float cx, cy, cz;
  __shared__ float sval[256];
  __shared__ int   sidx[256];

  int sel = 0;                                // reference starts at index 0
  for (int s = 0; s < npoint; ++s){
    if (tid == 0){
      cx = X[sel*3+0]; cy = X[sel*3+1]; cz = X[sel*3+2];
      float* o = new_xyz + ((long long)b * npoint + s) * 3;
      o[0] = cx; o[1] = cy; o[2] = cz;
    }
    __syncthreads();
    float bx = cx, by = cy, bz = cz;
    float best = -1.0f; int bi = 0;
#pragma unroll
    for (int i = 0; i < PPT; ++i){
      int j = tid + i * T;
      if (j < N){
        float dx = X[j*3+0]-bx, dy = X[j*3+1]-by, dz = X[j*3+2]-bz;
        float d = dx*dx + dy*dy + dz*dz;
        float nd = fminf(dist[i], d);
        dist[i] = nd;
        if (nd > best){ best = nd; bi = j; }
      }
    }
    sval[tid] = best; sidx[tid] = bi;
    __syncthreads();
    for (int off = T >> 1; off > 0; off >>= 1){
      if (tid < off){
        float v2 = sval[tid+off]; int i2 = sidx[tid+off];
        if (v2 > sval[tid] || (v2 == sval[tid] && i2 < sidx[tid])){
          sval[tid] = v2; sidx[tid] = i2;
        }
      }
      __syncthreads();
    }
    sel = sidx[0];
    __syncthreads();
  }
}

// ------------------------------------------------------------------
// Ball query: one wave32 per query, ordered slot assignment via
// ballot + prefix popcount.
// ------------------------------------------------------------------
__global__ void ballquery_kernel(const float* __restrict__ xyz,
                                 const float* __restrict__ qxyz,
                                 int* __restrict__ idx,
                                 int B, int N, int S, int ns, float r2){
  int gt   = blockIdx.x * blockDim.x + threadIdx.x;
  int wave = gt >> 5;
  int lane = threadIdx.x & 31;
  if (wave >= B * S) return;
  int b = wave / S, s = wave % S;
  const float* X = xyz + (long long)b * N * 3;
  const float* Q = qxyz + ((long long)b * S + s) * 3;
  float qx = Q[0], qy = Q[1], qz = Q[2];
  int* out = idx + (long long)wave * ns;

  int cnt = 0;
  int firstIdx = N - 1;                       // clamp behavior if nothing in range
  for (int base = 0; base < N && cnt < ns; base += 32){
    int j = base + lane;
    if (base + 32 < N)
      __builtin_prefetch(&X[(long long)(base + 32 + lane) * 3], 0, 0);
    bool within = false;
    if (j < N){
      float dx = X[j*3+0]-qx, dy = X[j*3+1]-qy, dz = X[j*3+2]-qz;
      within = (dx*dx + dy*dy + dz*dz) <= r2;
    }
    unsigned mask = (unsigned)__ballot(within);
    if (mask){
      if (cnt == 0) firstIdx = base + (__ffs(mask) - 1);
      int pre  = __popc(mask & ((1u << lane) - 1u));
      int slot = cnt + pre;
      if (within && slot < ns) out[slot] = j;
      cnt += __popc(mask);
    }
  }
  if (cnt > ns) cnt = ns;
  for (int k = cnt + lane; k < ns; k += 32) out[k] = firstIdx;
}

// ------------------------------------------------------------------
// Grouping kernels: emit bf16 rows, zero-padded to Kpad.
// ------------------------------------------------------------------
__global__ void group1_kernel(const float* __restrict__ xyz,
                              const float* __restrict__ qxyz,
                              const int* __restrict__ idx,
                              unsigned short* __restrict__ out,
                              int B, int N, int S, int ns, int Kpad){
  long long t = (long long)blockIdx.x * blockDim.x + threadIdx.x;
  long long total = (long long)B * S * ns * Kpad;
  if (t >= total) return;
  int c = (int)(t % Kpad);
  long long row = t / Kpad;
  long long bs  = row / ns;
  int s = (int)(bs % S), b = (int)(bs / S);
  unsigned short v = 0;
  if (c < 3){
    int j = idx[row];
    v = f2bf(xyz[((long long)b * N + j) * 3 + c] -
             qxyz[((long long)b * S + s) * 3 + c]);
  }
  out[t] = v;
}

__global__ void group2_kernel(const float* __restrict__ xyz,
                              const float* __restrict__ pts,
                              const float* __restrict__ qxyz,
                              const int* __restrict__ idx,
                              unsigned short* __restrict__ out,
                              int B, int N, int S, int ns, int C, int Kpad){
  long long t = (long long)blockIdx.x * blockDim.x + threadIdx.x;
  long long total = (long long)B * S * ns * Kpad;
  if (t >= total) return;
  int c = (int)(t % Kpad);
  long long row = t / Kpad;
  long long bs  = row / ns;
  int s = (int)(bs % S), b = (int)(bs / S);
  int j = idx[row];
  unsigned short v = 0;
  if (c < 3){
    v = f2bf(xyz[((long long)b * N + j) * 3 + c] -
             qxyz[((long long)b * S + s) * 3 + c]);
  } else if (c < 3 + C){
    v = f2bf(pts[((long long)b * N + j) * C + (c - 3)]);
  }
  out[t] = v;
}

__global__ void group3_kernel(const float* __restrict__ l2_xyz,
                              const float* __restrict__ l2_pts,
                              unsigned short* __restrict__ out,
                              int rows, int C, int Kpad){   // C = 256
  long long t = (long long)blockIdx.x * blockDim.x + threadIdx.x;
  long long total = (long long)rows * Kpad;
  if (t >= total) return;
  int c = (int)(t % Kpad);
  long long r = t / Kpad;
  unsigned short v = 0;
  if (c < 3)            v = f2bf(l2_xyz[r * 3 + c]);
  else if (c < 3 + C)   v = f2bf(l2_pts[r * (long long)C + (c - 3)]);
  out[t] = v;
}

// ------------------------------------------------------------------
// GEMM: C[m,n] = sum_k A[m,k]*W[n,k] + bias[n]; bf16 in, f32 acc.
// One wave32 computes a 16x64 strip. Per k-step, ALL 10 b128 loads
// (A + 4 B tiles, distinct live registers) are issued before the 4
// WMMAs so the compiler can use partial s_wait_loadcnt and keep the
// matrix pipe busy. Kpad % 32 == 0 -> no tails, 16B-aligned loads,
// wave-uniform guard keeps EXEC all-ones for every WMMA.
// ------------------------------------------------------------------
__global__ __launch_bounds__(256)
void gemm_bf16_wmma_kernel(const unsigned short* __restrict__ A,
                           const unsigned short* __restrict__ W,
                           const float* __restrict__ bias,
                           float* __restrict__ C,
                           int M, int N, int Kpad, int Mstore){
  int waveInBlock = threadIdx.x >> 5;
  int lane        = threadIdx.x & 31;
  int tile        = blockIdx.x * 8 + waveInBlock;
  int Mt = M >> 4, Nt4 = N >> 6;
  if (tile >= Mt * Nt4) return;               // wave-uniform guard
  int mt = tile % Mt, ntb = tile / Mt;

  int rA   = mt * 16 + (lane & 15);
  int rW0  = ntb * 64 + (lane & 15);
  int koff = (lane >> 4) ? 8 : 0;             // A/B lane K-halves (ISA layout)
  const unsigned short* pa = A + (long long)rA * Kpad + koff;
  const unsigned short* pw = W + (long long)rW0 * Kpad + koff;
  const long long wstride = (long long)16 * Kpad;

  union BfVec { v16bf v; uint4 q[2]; };
  v8f acc[4] = {{}, {}, {}, {}};
  for (int k0 = 0; k0 < Kpad; k0 += 32){
    BfVec a, b0, b1, b2, b3;
    // issue every load of this k-step before any WMMA consumes data
    a.q[0]  = *reinterpret_cast<const uint4*>(pa + k0);
    a.q[1]  = *reinterpret_cast<const uint4*>(pa + k0 + 16);
    b0.q[0] = *reinterpret_cast<const uint4*>(pw + 0 * wstride + k0);
    b0.q[1] = *reinterpret_cast<const uint4*>(pw + 0 * wstride + k0 + 16);
    b1.q[0] = *reinterpret_cast<const uint4*>(pw + 1 * wstride + k0);
    b1.q[1] = *reinterpret_cast<const uint4*>(pw + 1 * wstride + k0 + 16);
    b2.q[0] = *reinterpret_cast<const uint4*>(pw + 2 * wstride + k0);
    b2.q[1] = *reinterpret_cast<const uint4*>(pw + 2 * wstride + k0 + 16);
    b3.q[0] = *reinterpret_cast<const uint4*>(pw + 3 * wstride + k0);
    b3.q[1] = *reinterpret_cast<const uint4*>(pw + 3 * wstride + k0 + 16);
    acc[0] = __builtin_amdgcn_wmma_f32_16x16x32_bf16(
        false, a.v, false, b0.v, (short)0, acc[0], false, false);
    acc[1] = __builtin_amdgcn_wmma_f32_16x16x32_bf16(
        false, a.v, false, b1.v, (short)0, acc[1], false, false);
    acc[2] = __builtin_amdgcn_wmma_f32_16x16x32_bf16(
        false, a.v, false, b2.v, (short)0, acc[2], false, false);
    acc[3] = __builtin_amdgcn_wmma_f32_16x16x32_bf16(
        false, a.v, false, b3.v, (short)0, acc[3], false, false);
  }

  int mbase = mt * 16 + ((lane >> 4) << 3);
#pragma unroll
  for (int j = 0; j < 4; ++j){
    int n    = ntb * 64 + j * 16 + (lane & 15);
    float bv = bias ? bias[n] : 0.0f;
#pragma unroll
    for (int r = 0; r < 8; ++r){
      int m = mbase + r;
      if (m < Mstore) C[(long long)m * N + n] = acc[j][r] + bv;
    }
  }
}

// ------------------------------------------------------------------
// BN statistics: per-column sum / sumsq (register partials + atomics)
// ------------------------------------------------------------------
__global__ void bn_stats_kernel(const float* __restrict__ H,
                                float* __restrict__ stats, // [2*N]
                                int M, int N, int rowsPerBlock){
  int c = blockIdx.x * blockDim.x + threadIdx.x;
  if (c >= N) return;
  int r0 = blockIdx.y * rowsPerBlock;
  int r1 = min(r0 + rowsPerBlock, M);
  float s = 0.0f, sq = 0.0f;
  for (int r = r0; r < r1; ++r){
    float v = H[(long long)r * N + c];
    s += v; sq += v * v;
  }
  atomicAdd(&stats[c], s);
  atomicAdd(&stats[N + c], sq);
}

// ------------------------------------------------------------------
// BN + ReLU + max over neighborhood dim
// ------------------------------------------------------------------
__global__ void bn_relu_max_kernel(const float* __restrict__ H,
                                   const float* __restrict__ stats,
                                   const float* __restrict__ gamma,
                                   const float* __restrict__ beta,
                                   float* __restrict__ outp,
                                   int G, int Kg, int N, float invCnt){
  int t = blockIdx.x * blockDim.x + threadIdx.x;
  if (t >= G * N) return;
  int n = t % N, g = t / N;
  float mean  = stats[n] * invCnt;
  float var   = stats[N + n] * invCnt - mean * mean;
  float scale = gamma[n] * rsqrtf(var + 1e-5f);
  float shift = beta[n] - mean * scale;
  const float* h = H + (long long)g * Kg * N + n;
  float m = 0.0f;                              // relu floor
  for (int k = 0; k < Kg; ++k){
    float v = h[(long long)k * N] * scale + shift;
    m = fmaxf(m, fmaxf(v, 0.0f));
  }
  outp[t] = m;
}

// l3 (8x512 f32) -> 16x512 bf16, extra rows zeroed (GEMM M padding)
__global__ void pad_l3_kernel(const float* __restrict__ l3,
                              unsigned short* __restrict__ out){
  int t = blockIdx.x * blockDim.x + threadIdx.x;   // 16*512
  if (t >= 16 * 512) return;
  int r = t / 512, c = t % 512;
  out[t] = (r < 8) ? f2bf(l3[r * 512 + c]) : (unsigned short)0;
}

// ==================================================================
extern "C" void kernel_launch(void* const* d_in, const int* in_sizes, int n_in,
                              void* d_out, int out_size, void* d_ws, size_t ws_size,
                              hipStream_t stream){
  (void)in_sizes; (void)n_in; (void)out_size; (void)ws_size;
  const float* xyz = (const float*)d_in[0];
  const float* w1  = (const float*)d_in[1];
  const float* b1  = (const float*)d_in[2];
  const float* g1  = (const float*)d_in[3];
  const float* be1 = (const float*)d_in[4];
  const float* w2  = (const float*)d_in[5];
  const float* b2  = (const float*)d_in[6];
  const float* g2  = (const float*)d_in[7];
  const float* be2 = (const float*)d_in[8];
  const float* w3  = (const float*)d_in[9];
  const float* b3  = (const float*)d_in[10];
  const float* g3  = (const float*)d_in[11];
  const float* be3 = (const float*)d_in[12];
  const float* wf  = (const float*)d_in[13];
  const float* bf  = (const float*)d_in[14];
  float* outp = (float*)d_out;

  const int B = 8, N1 = 8192, S1 = 512, NS1 = 32;
  const int S2 = 128, NS2 = 64;
  const int K1p = 32, K2p = 160, K3p = 288, KFp = 512;

  // ---- bump allocator over d_ws ----
  char* base = (char*)d_ws;
  size_t off = 0;
  auto alloc = [&](size_t bytes) -> char* {
    char* p = base + off;
    off += (bytes + 255) & ~(size_t)255;
    return p;
  };
  unsigned short* w1b = (unsigned short*)alloc((size_t)128 * K1p * 2);
  unsigned short* w2b = (unsigned short*)alloc((size_t)256 * K2p * 2);
  unsigned short* w3b = (unsigned short*)alloc((size_t)512 * K3p * 2);
  unsigned short* wfb = (unsigned short*)alloc((size_t)256 * KFp * 2);
  float* l1_xyz = (float*)alloc((size_t)B * S1 * 3 * 4);
  int*   idx1   = (int*)  alloc((size_t)B * S1 * NS1 * 4);
  float* l1_pts = (float*)alloc((size_t)B * S1 * 128 * 4);
  float* l2_xyz = (float*)alloc((size_t)B * S2 * 3 * 4);
  int*   idx2   = (int*)  alloc((size_t)B * S2 * NS2 * 4);
  float* l2_pts = (float*)alloc((size_t)B * S2 * 256 * 4);
  unsigned short* g3buf = (unsigned short*)alloc((size_t)B * S2 * K3p * 2);
  float* h3   = (float*)alloc((size_t)B * S2 * 512 * 4);
  float* l3   = (float*)alloc((size_t)B * 512 * 4);
  unsigned short* l3b = (unsigned short*)alloc((size_t)16 * 512 * 2);
  float* stats = (float*)alloc((size_t)2 * 512 * 4);
  // reused big regions
  unsigned short* groupedB = (unsigned short*)alloc((size_t)B * S2 * NS2 * K2p * 2); // 20MB (>= group1's 8MB)
  float* hA = (float*)alloc((size_t)B * S1 * NS1 * 128 * 4);                         // 64MB (== h2 size)

  auto cdiv = [](long long a, long long b){ return (int)((a + b - 1) / b); };

  // ---- weight packing ----
  convert_pad_bf16_kernel<<<cdiv((long long)128*K1p,256),256,0,stream>>>(w1, w1b, 128, 3,   K1p);
  convert_pad_bf16_kernel<<<cdiv((long long)256*K2p,256),256,0,stream>>>(w2, w2b, 256, 131, K2p);
  convert_pad_bf16_kernel<<<cdiv((long long)512*K3p,256),256,0,stream>>>(w3, w3b, 512, 259, K3p);
  convert_pad_bf16_kernel<<<cdiv((long long)256*KFp,256),256,0,stream>>>(wf, wfb, 256, 512, KFp);

  // ================= Stage 1 =================
  fps_kernel<32><<<B,256,0,stream>>>(xyz, l1_xyz, N1, S1);
  ballquery_kernel<<<cdiv((long long)B*S1*32,256),256,0,stream>>>(
      xyz, l1_xyz, idx1, B, N1, S1, NS1, 0.04f);
  group1_kernel<<<cdiv((long long)B*S1*NS1*K1p,256),256,0,stream>>>(
      xyz, l1_xyz, idx1, groupedB, B, N1, S1, NS1, K1p);
  {
    int M = B * S1 * NS1;                       // 131072
    int tiles = (M/16) * (128/64);
    gemm_bf16_wmma_kernel<<<cdiv(tiles,8),256,0,stream>>>(
        groupedB, w1b, b1, hA, M, 128, K1p, M);
    zero_f32_kernel<<<1,256,0,stream>>>(stats, 2*128);
    bn_stats_kernel<<<dim3(1, cdiv(M,512)),128,0,stream>>>(hA, stats, M, 128, 512);
    bn_relu_max_kernel<<<cdiv((long long)B*S1*128,256),256,0,stream>>>(
        hA, stats, g1, be1, l1_pts, B*S1, NS1, 128, 1.0f/(float)M);
  }

  // ================= Stage 2 =================
  fps_kernel<2><<<B,256,0,stream>>>(l1_xyz, l2_xyz, S1, S2);
  ballquery_kernel<<<cdiv((long long)B*S2*32,256),256,0,stream>>>(
      l1_xyz, l2_xyz, idx2, B, S1, S2, NS2, 0.16f);
  group2_kernel<<<cdiv((long long)B*S2*NS2*K2p,256),256,0,stream>>>(
      l1_xyz, l1_pts, l2_xyz, idx2, groupedB, B, S1, S2, NS2, 128, K2p);
  {
    int M = B * S2 * NS2;                       // 65536
    int tiles = (M/16) * (256/64);
    gemm_bf16_wmma_kernel<<<cdiv(tiles,8),256,0,stream>>>(
        groupedB, w2b, b2, hA, M, 256, K2p, M);
    zero_f32_kernel<<<1,256,0,stream>>>(stats, 2*256);
    bn_stats_kernel<<<dim3(2, cdiv(M,512)),128,0,stream>>>(hA, stats, M, 256, 512);
    bn_relu_max_kernel<<<cdiv((long long)B*S2*256,256),256,0,stream>>>(
        hA, stats, g2, be2, l2_pts, B*S2, NS2, 256, 1.0f/(float)M);
  }

  // ================= Stage 3 (group-all) =================
  group3_kernel<<<cdiv((long long)B*S2*K3p,256),256,0,stream>>>(
      l2_xyz, l2_pts, g3buf, B*S2, 256, K3p);
  {
    int M = B * S2;                             // 1024
    int tiles = (M/16) * (512/64);
    gemm_bf16_wmma_kernel<<<cdiv(tiles,8),256,0,stream>>>(
        g3buf, w3b, b3, h3, M, 512, K3p, M);
    zero_f32_kernel<<<2,256,0,stream>>>(stats, 2*512);
    bn_stats_kernel<<<dim3(4, cdiv(M,512)),128,0,stream>>>(h3, stats, M, 512, 512);
    bn_relu_max_kernel<<<cdiv((long long)B*512,256),256,0,stream>>>(
        h3, stats, g3, be3, l3, B, S2, 512, 1.0f/(float)M);
  }

  // ================= Head: out = l3 @ wf^T + bf =================
  pad_l3_kernel<<<cdiv(16*512,256),256,0,stream>>>(l3, l3b);
  {
    int tiles = (16/16) * (256/64);             // 4 tiles -> 1 block, uniform guard
    gemm_bf16_wmma_kernel<<<cdiv(tiles,8),256,0,stream>>>(
        l3b, wfb, bf, outp, 16, 256, KFp, 8);
  }
}
